// attentionBlock_88141318848709
// MI455X (gfx1250) — matrix operand order repaired and checked by reading, not compiled, not run on hardware
//
#include <hip/hip_runtime.h>
#include <hip/hip_bf16.h>
#include <math.h>
#include <stdint.h>

// ---------------------------------------------------------------------------
// Problem constants (from reference)
// ---------------------------------------------------------------------------
constexpr int Bc = 4, Tc = 2048, Cc = 1024, Hc = 16, Dc = 64;
constexpr int Mc = Bc * Tc;          // 8192 rows
constexpr int CTX = 128;

typedef __bf16  v16bf __attribute__((ext_vector_type(16)));
typedef __bf16  v8bf  __attribute__((ext_vector_type(8)));
typedef float   v8f   __attribute__((ext_vector_type(8)));

// ---------------------------------------------------------------------------
// WMMA helper: D = A(16x32 bf16) x B(32x16 bf16) + C(f32)
// ---------------------------------------------------------------------------
static __device__ __forceinline__ v8f wmma_bf16(v16bf a, v16bf b, v8f c) {
    return __builtin_amdgcn_wmma_f32_16x16x32_bf16(
        /*neg_a=*/false, a, /*neg_b=*/false, b,
        /*c_mod=*/(short)0, c, /*reuse_a=*/false, /*reuse_b=*/false);
}

static __device__ __forceinline__ v16bf cat8(v8bf a, v8bf b) {
    return __builtin_shufflevector(a, b, 0,1,2,3,4,5,6,7,8,9,10,11,12,13,14,15);
}

// A-matrix 16x32 bf16 fragment (this code's k order, same as B): lane L holds row M=L&15.
// halves e -> K = kb + e, kb=(L<16)?0:16 (contiguous 16 halves)
static __device__ __forceinline__ v16bf load_frag_a(const __bf16* base, int row,
                                                    int stride, int kofs) {
    const int lane = threadIdx.x & 31;
    const int r  = row + (lane & 15);
    const int kb = (lane < 16) ? 0 : 16;
    const __bf16* p = base + r * stride + kofs + kb;
    v8bf lo = *(const v8bf*)(p);
    v8bf hi = *(const v8bf*)(p + 8);
    return cat8(lo, hi);
}

// B-matrix 32x16 bf16 fragment: lane L holds column N=L&15,
// halves e -> K = klo + e, klo = (L<16)?0:16 (contiguous 16 halves).
static __device__ __forceinline__ v16bf load_frag_b(const __bf16* base, int col,
                                                    int stride, int kofs) {
    const int lane = threadIdx.x & 31;
    const int n   = col + (lane & 15);
    const int klo = (lane < 16) ? 0 : 16;
    const __bf16* p = base + n * stride + kofs + klo;
    v8bf lo = *(const v8bf*)(p);
    v8bf hi = *(const v8bf*)(p + 8);
    return cat8(lo, hi);
}

// ---------------------------------------------------------------------------
// CDNA5 async global -> LDS (ASYNCcnt-tracked DMA), GVS addressing:
// 16 bytes from global saddr + vaddr + offset land in LDS at vdst + offset
// ---------------------------------------------------------------------------
static __device__ __forceinline__ void async_b128(uint32_t lds_off,
                                                  uint64_t gbase, int32_t voff) {
    asm volatile("global_load_async_to_lds_b128 %0, %1, %2"
                 :: "v"(lds_off), "v"(voff), "s"(gbase) : "memory");
}
static __device__ __forceinline__ void async_b128_o16(uint32_t lds_off,
                                                      uint64_t gbase, int32_t voff) {
    asm volatile("global_load_async_to_lds_b128 %0, %1, %2 offset:16"
                 :: "v"(lds_off), "v"(voff), "s"(gbase) : "memory");
}
static __device__ __forceinline__ void wait_async0() {
    asm volatile("s_wait_asynccnt 0x0" ::: "memory");
}

// ---------------------------------------------------------------------------
// f32 -> bf16 conversion
// ---------------------------------------------------------------------------
__global__ void cvt_bf16_kernel(const float* __restrict__ s,
                                __bf16* __restrict__ d, int n) {
    int i = blockIdx.x * blockDim.x + threadIdx.x;
    if (i < n) d[i] = (__bf16)s[i];
}

// ---------------------------------------------------------------------------
// GEMM: Y[m,n] = act( sum_k A[m,k] * W[n,k] + bias[n] )
// A: [Mc, Cc] bf16, W: [Cc, Cc] bf16 (row = output channel), Y: [Mc, Cc] bf16
// Workgroup tile 128x128, 8 waves (32x64 each, 2x4 WMMA tiles), K-step 32.
// Tiles staged via async global->LDS DMA with double-buffered LDS.
// ---------------------------------------------------------------------------
__global__ __launch_bounds__(256) void gemm_bf16_kernel(
    const __bf16* __restrict__ A, const __bf16* __restrict__ W,
    const float* __restrict__ bias, __bf16* __restrict__ Y, int act) {

    __shared__ __bf16 As[2][128 * 32];
    __shared__ __bf16 Bs[2][128 * 32];

    const int tid  = threadIdx.x;
    const int lane = tid & 31;
    const int wid  = tid >> 5;
    const int wm   = wid & 3;       // 4 waves along M (32 rows each)
    const int wn   = wid >> 2;      // 2 waves along N (64 cols each)
    const int row0 = blockIdx.y * 128;
    const int col0 = blockIdx.x * 128;

    v8f acc[2][4];
    for (int a = 0; a < 2; ++a)
        for (int b = 0; b < 4; ++b)
            acc[a][b] = v8f{0.f, 0.f, 0.f, 0.f, 0.f, 0.f, 0.f, 0.f};

    // cooperative async-copy assignment: thread -> (row, 16-half chunk)
    const int ldrow = tid >> 1;               // 0..127
    const int ldoff = (tid & 1) * 16;         // halves
    const int32_t a_voff = ((row0 + ldrow) * Cc + ldoff) * 2;  // bytes
    const int32_t b_voff = ((col0 + ldrow) * Cc + ldoff) * 2;  // bytes
    const uint64_t a_gb = (uint64_t)(uintptr_t)A;
    const uint64_t b_gb = (uint64_t)(uintptr_t)W;
    uint32_t a_lds[2], b_lds[2];
    for (int p = 0; p < 2; ++p) {
        a_lds[p] = (uint32_t)(uintptr_t)(&As[p][ldrow * 32 + ldoff]);
        b_lds[p] = (uint32_t)(uintptr_t)(&Bs[p][ldrow * 32 + ldoff]);
    }

    // prologue: fill buffer 0 for kk = 0
    async_b128   (a_lds[0], a_gb, a_voff);
    async_b128_o16(a_lds[0], a_gb, a_voff);
    async_b128   (b_lds[0], b_gb, b_voff);
    async_b128_o16(b_lds[0], b_gb, b_voff);

    int p = 0;
    for (int kk = 0; kk < Cc; kk += 32) {
        wait_async0();        // this wave's current-buffer DMAs have landed
        __syncthreads();      // all waves landed + all done reading other buf
        if (kk + 32 < Cc) {   // prefetch next K-slab into the other buffer
            const int np = p ^ 1;
            const int32_t kb = (kk + 32) * 2;
            async_b128   (a_lds[np], a_gb, a_voff + kb);
            async_b128_o16(a_lds[np], a_gb, a_voff + kb);
            async_b128   (b_lds[np], b_gb, b_voff + kb);
            async_b128_o16(b_lds[np], b_gb, b_voff + kb);
        }
        v16bf af0 = load_frag_a(As[p], wm * 32,      32, 0);
        v16bf af1 = load_frag_a(As[p], wm * 32 + 16, 32, 0);
        for (int tn = 0; tn < 4; ++tn) {
            v16bf bf = load_frag_b(Bs[p], wn * 64 + tn * 16, 32, 0);
            acc[0][tn] = wmma_bf16(af0, bf, acc[0][tn]);
            acc[1][tn] = wmma_bf16(af1, bf, acc[1][tn]);
        }
        p ^= 1;
    }

    const int rsel = (lane < 16) ? 0 : 8;
    for (int tm = 0; tm < 2; ++tm)
        for (int tn = 0; tn < 4; ++tn) {
            const int n = col0 + wn * 64 + tn * 16 + (lane & 15);
            const float bv = bias[n];
            for (int i = 0; i < 8; ++i) {
                const int m = row0 + wm * 32 + tm * 16 + i + rsel;
                float v = acc[tm][tn][i] + bv;
                if (act == 1) v = (v > 0.f) ? v : (__expf(v) - 1.f);  // ELU
                Y[(size_t)m * Cc + n] = (__bf16)v;
            }
        }
}

// ---------------------------------------------------------------------------
// RoPE on interleaved pairs, in place on bf16 q and k.
// ---------------------------------------------------------------------------
__global__ void rope_kernel(__bf16* __restrict__ q, __bf16* __restrict__ k,
                            int npairs) {
    int idx = blockIdx.x * blockDim.x + threadIdx.x;
    if (idx >= npairs) return;
    const int c2 = idx % (Cc / 2);
    const int t  = (idx / (Cc / 2)) % Tc;
    const int d  = (c2 * 2) % Dc;        // position within head
    const int dp = d >> 1;
    const float theta = __expf(-9.2103403719761836f * (float)(2 * dp) / 64.f);
    const float ang = (float)t * theta;
    float s, c;
    __sincosf(ang, &s, &c);
    const size_t base = (size_t)idx * 2;
    float x0 = (float)q[base], x1 = (float)q[base + 1];
    q[base]     = (__bf16)(c * x0 - s * x1);
    q[base + 1] = (__bf16)(s * x0 + c * x1);
    x0 = (float)k[base]; x1 = (float)k[base + 1];
    k[base]     = (__bf16)(c * x0 - s * x1);
    k[base + 1] = (__bf16)(s * x0 + c * x1);
}

// ---------------------------------------------------------------------------
// Banded flash attention. One wave per (b, h, 16-query tile).
// Computes S^T = K(16x64) x Q^T(64x16) so the per-query softmax reduction is
// a single shfl_xor(16); online softmax over 32-key blocks; P round-trips
// through LDS into A-fragment layout; V stored transposed in LDS for PV.
// ---------------------------------------------------------------------------
__global__ __launch_bounds__(32) void attn_kernel(
    const __bf16* __restrict__ q, const __bf16* __restrict__ k,
    const __bf16* __restrict__ v, const int* __restrict__ x_len,
    __bf16* __restrict__ y) {

    constexpr int KL = 72;   // k_lds row stride (halves), 16B aligned, bank-spread
    constexpr int VL = 40;   // vT_lds row stride
    constexpr int PL = 40;   // p_lds row stride
    __shared__ __bf16 k_lds[32 * KL];
    __shared__ __bf16 vT_lds[64 * VL];
    __shared__ __bf16 p_lds[16 * PL];

    const int lane = threadIdx.x;
    const int t0 = blockIdx.x * 16;
    const int h  = blockIdx.y;
    const int b  = blockIdx.z;
    const int len = x_len[b];

    const size_t head_off = ((size_t)b * Tc) * Cc + (size_t)h * Dc;
    const __bf16* qbase = q + head_off + (size_t)t0 * Cc;

    v16bf qf0 = load_frag_b(qbase, 0, Cc, 0);    // Q^T, d = 0..31
    v16bf qf1 = load_frag_b(qbase, 0, Cc, 32);   // Q^T, d = 32..63

    v8f o[4];
    for (int dt = 0; dt < 4; ++dt) o[dt] = v8f{0.f,0.f,0.f,0.f,0.f,0.f,0.f,0.f};
    float mstat = -1e30f, lstat = 0.f;

    const int qcol = lane & 15;
    const int hh8  = (lane < 16) ? 0 : 8;
    const int qt   = t0 + qcol;

    const int slo = (t0 > CTX) ? ((t0 - CTX) & ~31) : 0;
    const int shi = (Tc < t0 + 16 + CTX) ? Tc : (t0 + 16 + CTX);

    for (int sb = slo; sb < shi; sb += 32) {
        __syncthreads();
        // cooperative load: lane -> key row sb+lane (clamped; masked later)
        const int srow = sb + lane;
        const int srowc = (srow < Tc) ? srow : (Tc - 1);
        const __bf16* kr = k + head_off + (size_t)srowc * Cc;
        const __bf16* vr = v + head_off + (size_t)srowc * Cc;
        for (int j = 0; j < 8; ++j) {
            v8bf kv = *(const v8bf*)(kr + j * 8);
            *(v8bf*)(&k_lds[lane * KL + j * 8]) = kv;
        }
        for (int j = 0; j < 8; ++j) {
            v8bf vv = *(const v8bf*)(vr + j * 8);
            for (int e = 0; e < 8; ++e)
                vT_lds[(j * 8 + e) * VL + lane] = vv[e];   // transpose store
        }
        __syncthreads();

        // scores: two 16-key tiles, S^T layout (rows = keys, cols = queries)
        v8f st[2];
        float tmax = -1e30f;
        for (int hb = 0; hb < 2; ++hb) {
            v8f sc = v8f{0.f,0.f,0.f,0.f,0.f,0.f,0.f,0.f};
            v16bf a0 = load_frag_a(k_lds, hb * 16, KL, 0);
            v16bf a1 = load_frag_a(k_lds, hb * 16, KL, 32);
            sc = wmma_bf16(a0, qf0, sc);
            sc = wmma_bf16(a1, qf1, sc);
            for (int i = 0; i < 8; ++i) {
                const int j = sb + hb * 16 + i + hh8;
                const int dd = (j > qt) ? (j - qt) : (qt - j);
                const bool ok = (j < len) && (qt < len) && (dd <= CTX);
                const float sv = ok ? sc[i] * 0.125f : -1e10f;  // 1/sqrt(64)
                sc[i] = sv;
                tmax = fmaxf(tmax, sv);
            }
            st[hb] = sc;
        }
        const float cmax  = fmaxf(tmax, __shfl_xor(tmax, 16));
        const float m_new = fmaxf(mstat, cmax);
        const float resc  = __expf(mstat - m_new);
        float rsum = 0.f;
        for (int hb = 0; hb < 2; ++hb)
            for (int i = 0; i < 8; ++i) {
                const float p = __expf(st[hb][i] - m_new);
                rsum += p;
                p_lds[qcol * PL + hb * 16 + i + hh8] = (__bf16)p;  // P^T -> P
            }
        rsum += __shfl_xor(rsum, 16);
        lstat = lstat * resc + rsum;
        mstat = m_new;

        // rescale accumulators (o rows are queries i+hh8; stats live at lane q)
        float f[8];
        for (int i = 0; i < 8; ++i) f[i] = __shfl(resc, i + hh8);
        for (int dt = 0; dt < 4; ++dt)
            for (int i = 0; i < 8; ++i) o[dt][i] *= f[i];

        __syncthreads();
        v16bf pf = load_frag_a(p_lds, 0, PL, 0);          // P as A (K = 32 keys)
        for (int dt = 0; dt < 4; ++dt) {
            v16bf bv = load_frag_b(vT_lds, dt * 16, VL, 0);
            o[dt] = wmma_bf16(pf, bv, o[dt]);
        }
    }

    // finalize: divide by row-sum, write y[b, t, h*64 + d]
    for (int i = 0; i < 8; ++i) {
        const float lq  = __shfl(lstat, i + hh8);
        const float inv = (lq > 0.f) ? (1.f / lq) : 0.f;
        const int qrow = t0 + i + hh8;
        for (int dt = 0; dt < 4; ++dt) {
            const float val = o[dt][i] * inv;
            y[head_off + (size_t)qrow * Cc + dt * 16 + qcol] = (__bf16)val;
        }
    }
}

// ---------------------------------------------------------------------------
// LayerNorm over rows of (a_f32 + b_bf16); writes f32 and/or bf16 outputs.
// ---------------------------------------------------------------------------
__global__ __launch_bounds__(256) void ln_kernel(
    const float* __restrict__ a, const __bf16* __restrict__ bb,
    const float* __restrict__ g, const float* __restrict__ be,
    float* __restrict__ o32, __bf16* __restrict__ o16) {

    const int row = blockIdx.x;
    const int tid = threadIdx.x;
    __shared__ float red[8];

    float vals[4];
    float s = 0.f;
    for (int j = 0; j < 4; ++j) {
        const int c = j * 256 + tid;
        const float x = a[(size_t)row * Cc + c] + (float)bb[(size_t)row * Cc + c];
        vals[j] = x;
        s += x;
    }
    for (int off = 16; off; off >>= 1) s += __shfl_down(s, off);
    if ((tid & 31) == 0) red[tid >> 5] = s;
    __syncthreads();
    if (tid < 8) {
        float t = red[tid];
        t += __shfl_down(t, 4); t += __shfl_down(t, 2); t += __shfl_down(t, 1);
        if (tid == 0) red[0] = t;
    }
    __syncthreads();
    const float mean = red[0] * (1.f / Cc);
    __syncthreads();

    float vs = 0.f;
    for (int j = 0; j < 4; ++j) {
        const float d = vals[j] - mean;
        vs += d * d;
    }
    for (int off = 16; off; off >>= 1) vs += __shfl_down(vs, off);
    if ((tid & 31) == 0) red[tid >> 5] = vs;
    __syncthreads();
    if (tid < 8) {
        float t = red[tid];
        t += __shfl_down(t, 4); t += __shfl_down(t, 2); t += __shfl_down(t, 1);
        if (tid == 0) red[0] = t;
    }
    __syncthreads();
    const float var  = red[0] * (1.f / Cc);
    const float rstd = rsqrtf(var + 1e-5f);

    for (int j = 0; j < 4; ++j) {
        const int c = j * 256 + tid;
        const float yv = (vals[j] - mean) * rstd * g[c] + be[c];
        if (o32) o32[(size_t)row * Cc + c] = yv;
        if (o16) o16[(size_t)row * Cc + c] = (__bf16)yv;
    }
}

// ---------------------------------------------------------------------------
// Host launch
// ---------------------------------------------------------------------------
extern "C" void kernel_launch(void* const* d_in, const int* in_sizes, int n_in,
                              void* d_out, int out_size, void* d_ws, size_t ws_size,
                              hipStream_t stream) {
    (void)in_sizes; (void)n_in; (void)out_size; (void)ws_size;

    const float* x   = (const float*)d_in[0];
    const float* Wq  = (const float*)d_in[1];  const float* bq = (const float*)d_in[2];
    const float* Wk  = (const float*)d_in[3];  const float* bk = (const float*)d_in[4];
    const float* Wv  = (const float*)d_in[5];  const float* bv = (const float*)d_in[6];
    const float* Wo  = (const float*)d_in[7];  const float* bo = (const float*)d_in[8];
    const float* W1  = (const float*)d_in[9];  const float* b1 = (const float*)d_in[10];
    const float* W2  = (const float*)d_in[11]; const float* b2 = (const float*)d_in[12];
    const float* g1  = (const float*)d_in[13]; const float* be1 = (const float*)d_in[14];
    const float* g2  = (const float*)d_in[15]; const float* be2 = (const float*)d_in[16];
    const int*   x_len = (const int*)d_in[17];
    float* out = (float*)d_out;

    char* ws = (char*)d_ws;
    const size_t actB = (size_t)Mc * Cc * sizeof(__bf16);   // 16 MiB
    const size_t wB   = (size_t)Cc * Cc * sizeof(__bf16);   //  2 MiB
    __bf16* xb  = (__bf16*)(ws);                       // x bf16; later y (attn out)
    __bf16* wqb = (__bf16*)(ws + actB);
    __bf16* wkb = (__bf16*)(ws + actB + 1 * wB);
    __bf16* wvb = (__bf16*)(ws + actB + 2 * wB);
    __bf16* wob = (__bf16*)(ws + actB + 3 * wB);
    __bf16* w1b = (__bf16*)(ws + actB + 4 * wB);
    __bf16* w2b = (__bf16*)(ws + actB + 5 * wB);
    __bf16* qb  = (__bf16*)(ws + actB + 6 * wB);            // later attn-proj, ff
    __bf16* kb  = (__bf16*)(ws + 2 * actB + 6 * wB);        // later add_norm bf16
    __bf16* vb  = (__bf16*)(ws + 3 * actB + 6 * wB);        // later h (ELU out)
    float*  an32 = (float*)(ws + 4 * actB + 6 * wB);        // add_norm f32, 32 MiB

    const int nAct = Mc * Cc;         // 8,388,608
    const int nW   = Cc * Cc;         // 1,048,576

    // 1) bf16 copies
    cvt_bf16_kernel<<<(nAct + 255) / 256, 256, 0, stream>>>(x,  xb,  nAct);
    cvt_bf16_kernel<<<(nW + 255) / 256, 256, 0, stream>>>(Wq, wqb, nW);
    cvt_bf16_kernel<<<(nW + 255) / 256, 256, 0, stream>>>(Wk, wkb, nW);
    cvt_bf16_kernel<<<(nW + 255) / 256, 256, 0, stream>>>(Wv, wvb, nW);
    cvt_bf16_kernel<<<(nW + 255) / 256, 256, 0, stream>>>(Wo, wob, nW);
    cvt_bf16_kernel<<<(nW + 255) / 256, 256, 0, stream>>>(W1, w1b, nW);
    cvt_bf16_kernel<<<(nW + 255) / 256, 256, 0, stream>>>(W2, w2b, nW);

    const dim3 gemmGrid(Cc / 128, Mc / 128);   // (8, 64)

    // 2) Q/K/V projections
    gemm_bf16_kernel<<<gemmGrid, 256, 0, stream>>>(xb, wqb, bq, qb, 0);
    gemm_bf16_kernel<<<gemmGrid, 256, 0, stream>>>(xb, wkb, bk, kb, 0);
    gemm_bf16_kernel<<<gemmGrid, 256, 0, stream>>>(xb, wvb, bv, vb, 0);

    // 3) RoPE on q, k
    const int npairs = nAct / 2;
    rope_kernel<<<(npairs + 255) / 256, 256, 0, stream>>>(qb, kb, npairs);

    // 4) banded attention -> y (reuse xb)
    attn_kernel<<<dim3(Tc / 16, Hc, Bc), 32, 0, stream>>>(qb, kb, vb, x_len, xb);

    // 5) output projection -> qb (q no longer needed)
    gemm_bf16_kernel<<<gemmGrid, 256, 0, stream>>>(xb, wob, bo, qb, 0);

    // 6) add_norm = LN(x + attn_out): f32 -> an32, bf16 -> kb
    ln_kernel<<<Mc, 256, 0, stream>>>(x, qb, g1, be1, an32, kb);

    // 7) h = ELU(add_norm @ W1^T + b1) -> vb
    gemm_bf16_kernel<<<gemmGrid, 256, 0, stream>>>(kb, w1b, b1, vb, 1);

    // 8) ff = h @ W2^T + b2 -> qb
    gemm_bf16_kernel<<<gemmGrid, 256, 0, stream>>>(vb, w2b, b2, qb, 0);

    // 9) out = LN(ff + add_norm) -> d_out (f32)
    ln_kernel<<<Mc, 256, 0, stream>>>(an32, qb, g2, be2, out, (__bf16*)nullptr);
}
